// Decoder_76201309766417
// MI455X (gfx1250) — matrix operand order repaired and checked by reading, not compiled, not run on hardware
//
#include <hip/hip_runtime.h>

// ---------------------------------------------------------------------------
// CDNA5 (gfx1250) transformer decoder layer.
//  - All GEMM-shaped math: v_wmma_f32_16x16x32_f16 (wave32, f32 accumulate)
//  - Global->LDS staging: global_load_async_to_lds_b128 (ASYNCcnt-tracked DMA)
//    with double-buffered LDS tiles; the GEMM hot loop is branch-free with the
//    last K tile peeled, so steady state = issue async -> wait(3) -> barrier
//    -> ds_load_b128 fragments -> 4x WMMA -> barrier.
//  - Attention: flash-attention style, one wave per (b, h, 16-row q tile).
// ---------------------------------------------------------------------------

typedef _Float16 v16h __attribute__((ext_vector_type(16)));
typedef _Float16 h8   __attribute__((ext_vector_type(8)));
typedef float    v8f  __attribute__((ext_vector_type(8)));

#define WMMA_F16(a, b, c) \
    __builtin_amdgcn_wmma_f32_16x16x32_f16(false, (a), false, (b), (short)0, (c), false, false)

// Async DMA: 16 bytes per lane, global -> LDS, no VGPR round-trip.
// LDS address = low 32 bits of the generic shared pointer (aperture scheme).
#define ASYNC_B128(ldsp, gp)                                           \
    asm volatile("global_load_async_to_lds_b128 %0, %1, off"           \
                 :: "v"((unsigned int)(unsigned long long)(ldsp)),     \
                    "v"(gp)                                            \
                 : "memory")

#if defined(__has_builtin)
#if __has_builtin(__builtin_amdgcn_s_wait_asynccnt)
#define WAIT_ASYNC(n) __builtin_amdgcn_s_wait_asynccnt((n))
#endif
#endif
#ifndef WAIT_ASYNC
#define WAIT_ASYNC(n) asm volatile("s_wait_asynccnt %0" :: "n"(n) : "memory")
#endif

// ---------------------------------------------------------------------------
// fp32 -> fp16 cast
// ---------------------------------------------------------------------------
__global__ __launch_bounds__(256) void cast_f16_k(const float* __restrict__ in,
                                                  _Float16* __restrict__ out, int n) {
    int i = blockIdx.x * 256 + threadIdx.x;
    if (i < n) out[i] = (_Float16)in[i];
}

// ---------------------------------------------------------------------------
// W[K][N] fp32  ->  Wt[N][K] fp16   (so GEMM B-fragments read contiguous K)
// ---------------------------------------------------------------------------
__global__ __launch_bounds__(256) void transpose_cast_k(const float* __restrict__ W,
                                                        _Float16* __restrict__ Wt,
                                                        int K, int N) {
    __shared__ _Float16 tile[32][33];
    const int tx = threadIdx.x & 31, ty = threadIdx.x >> 5;          // 32 x 8
    const int k0 = blockIdx.y * 32, n0 = blockIdx.x * 32;
#pragma unroll
    for (int i = 0; i < 4; ++i)
        tile[ty + 8 * i][tx] = (_Float16)W[(size_t)(k0 + ty + 8 * i) * N + n0 + tx];
    __syncthreads();
#pragma unroll
    for (int i = 0; i < 4; ++i)
        Wt[(size_t)(n0 + ty + 8 * i) * K + k0 + tx] = tile[tx][ty + 8 * i];
}

// ---------------------------------------------------------------------------
// WMMA GEMM:  out = act((A @ W + bias) * scale) + resid
//   A : [M][K] f16 row-major,  Bt : [N][K] f16 (pre-transposed weight)
//   Block tile 128x64, 8 waves: wave w -> rows [w*16, w*16+16), all 64 cols.
// ---------------------------------------------------------------------------
#define BM 128
#define BN 64
#define BK 32

__global__ __launch_bounds__(256) void gemm_wmma_k(
    const _Float16* __restrict__ A, const _Float16* __restrict__ Bt,
    const float* __restrict__ bias, const float* __restrict__ resid,
    float* __restrict__ outF, _Float16* __restrict__ outH,
    int M, int K, int N, float scale, int act) {
    __shared__ __align__(64) _Float16 sA[2][BM][BK];
    __shared__ __align__(64) _Float16 sB[2][BN][BK];

    const int tid  = threadIdx.x;
    const int wave = tid >> 5;
    const int lane = tid & 31;
    const int half = lane >> 4;
    const int l15  = lane & 15;
    const int m0   = blockIdx.y * BM;
    const int n0   = blockIdx.x * BN;

    // Per-thread staging coordinates (16B chunks).
    const int ra = tid >> 2;             // A rows ra and ra+64
    const int ca = (tid & 3) * 8;        // column (halves)
    const int rb = tid >> 2;             // B row
    const int cb = (tid & 3) * 8;

    const _Float16* gA0 = &A[(size_t)(m0 + ra) * K + ca];
    const _Float16* gA1 = &A[(size_t)(m0 + ra + 64) * K + ca];
    const _Float16* gB  = &Bt[(size_t)(n0 + rb) * K + cb];

    // Prologue: stage K-tile 0 into buffer 0.
    ASYNC_B128(&sA[0][ra][ca],      gA0);
    ASYNC_B128(&sA[0][ra + 64][ca], gA1);
    ASYNC_B128(&sB[0][rb][cb],      gB);

    v8f acc[4] = {};
    int buf = 0;

    auto mma_step = [&](int bsel) {
        // A fragment (16x32): lane l elem j -> m=l&15, k=(l>>4)*16+j
        v16h af;
        ((h8*)&af)[0] = *(const h8*)&sA[bsel][wave * 16 + l15][half * 16];
        ((h8*)&af)[1] = *(const h8*)&sA[bsel][wave * 16 + l15][half * 16 + 8];
#pragma unroll
        for (int t = 0; t < 4; ++t) {
            // B fragment (32x16): lane l elem j -> n=l&15, k=(l>>4)*16+j
            v16h bf = *(const v16h*)&sB[bsel][t * 16 + l15][half * 16];
            acc[t] = WMMA_F16(af, bf, acc[t]);
        }
    };

    // Branch-free hot loop: always one tile in flight (last tile peeled).
    for (int k0 = 0; k0 < K - BK; k0 += BK) {
        const int kn = k0 + BK;
        ASYNC_B128(&sA[buf ^ 1][ra][ca],      gA0 + kn);
        ASYNC_B128(&sA[buf ^ 1][ra + 64][ca], gA1 + kn);
        ASYNC_B128(&sB[buf ^ 1][rb][cb],      gB + kn);
        WAIT_ASYNC(3);  // async loads complete in order: oldest 3 done
        __syncthreads();
        mma_step(buf);
        __syncthreads();  // everyone done reading before buffer is overwritten
        buf ^= 1;
    }
    // Peeled final tile.
    WAIT_ASYNC(0);
    __syncthreads();
    mma_step(buf);

    // Epilogue. C/D layout: lane l reg r -> m=(l>>4)*8+r, n=l&15
#pragma unroll
    for (int t = 0; t < 4; ++t) {
#pragma unroll
        for (int r = 0; r < 8; ++r) {
            int row = m0 + wave * 16 + half * 8 + r;
            int col = n0 + t * 16 + l15;
            float v = acc[t][r];
            if (bias) v += bias[col];
            v *= scale;
            if (act) v = 0.5f * v * (1.0f + erff(v * 0.70710678118654752f)); // exact GELU
            if (resid) v += resid[(size_t)row * N + col];
            if (outF) outF[(size_t)row * N + col] = v;
            if (outH) outH[(size_t)row * N + col] = (_Float16)v;
        }
    }
}

// ---------------------------------------------------------------------------
// Flash attention: one wave per (b, h, 16-row query tile). DH = 64.
// Q/K/V are [B*N][H*64] f16 (packed heads). Softmax scale pre-folded into Q.
// K block staged with async DMA (overlaps the V-transpose staging work).
// ---------------------------------------------------------------------------
__global__ __launch_bounds__(32) void attn_wmma_k(
    const _Float16* __restrict__ Q, const _Float16* __restrict__ Kh,
    const _Float16* __restrict__ Vh, _Float16* __restrict__ O,
    int Nn, int H, int causal) {
    const int lane = threadIdx.x;
    const int half = lane >> 4, l15 = lane & 15;
    int bid = blockIdx.x;
    const int qt = bid % (Nn / 16); bid /= (Nn / 16);
    const int h  = bid % H;
    const int b  = bid / H;
    const int q0 = qt * 16;
    const int E  = H * 64;
    const size_t base = ((size_t)b * Nn) * E + (size_t)h * 64;

    __shared__ __align__(64) _Float16 sK[32][64];   // [key][d]   (natural)
    __shared__ __align__(64) _Float16 sVt[64][32];  // [d][key]   (transposed)
    __shared__ __align__(64) _Float16 sP[16][32];   // probs 16q x 32k

    // Q fragments (d 0..31 and 32..63) straight from global: chunks already
    // match the A-fragment layout.
    v16h qf0, qf1;
    {
        const _Float16* qrow = Q + base + (size_t)(q0 + l15) * E;
        ((h8*)&qf0)[0] = *(const h8*)&qrow[half * 16];
        ((h8*)&qf0)[1] = *(const h8*)&qrow[half * 16 + 8];
        ((h8*)&qf1)[0] = *(const h8*)&qrow[32 + half * 16];
        ((h8*)&qf1)[1] = *(const h8*)&qrow[40 + half * 16];
    }

    v8f o[4] = {};
    float mi[8], li[8];
#pragma unroll
    for (int r = 0; r < 8; ++r) { mi[r] = -3.0e38f; li[r] = 0.0f; }

    const int kend = causal ? (q0 + 16) : Nn;
    for (int kb = 0; kb < kend; kb += 32) {
        // Issue async DMA for the K block [32][64]: 256 x 16B chunks.
#pragma unroll
        for (int i = 0; i < 8; ++i) {
            int c = lane + i * 32;
            int key = c >> 3, cc = (c & 7) * 8;
            ASYNC_B128(&sK[key][cc], &Kh[base + (size_t)(kb + key) * E + cc]);
        }
        // Meanwhile stage V transposed [64][32]: lane owns one key row.
        {
            const _Float16* vrow = Vh + base + (size_t)(kb + lane) * E;
#pragma unroll
            for (int i = 0; i < 8; ++i) {
                h8 v = *(const h8*)&vrow[i * 8];
#pragma unroll
                for (int j = 0; j < 8; ++j) sVt[i * 8 + j][lane] = v[j];
            }
        }
        WAIT_ASYNC(0);
        __syncthreads();

        // Scores for two 16-key subtiles: S = Q @ K^T (scale folded into Q).
        v8f sc[2];
#pragma unroll
        for (int s = 0; s < 2; ++s) {
            v16h b0 = *(const v16h*)&sK[s * 16 + l15][half * 16];
            v16h b1 = *(const v16h*)&sK[s * 16 + l15][32 + half * 16];
            v8f z = {};
            z = WMMA_F16(qf0, b0, z);
            z = WMMA_F16(qf1, b1, z);
            sc[s] = z;
        }
        if (causal) {
#pragma unroll
            for (int s = 0; s < 2; ++s) {
                int key = kb + s * 16 + l15;
#pragma unroll
                for (int r = 0; r < 8; ++r)
                    if (key > q0 + half * 8 + r) sc[s][r] = -3.0e38f;
            }
        }

        // Online softmax. Row m lives in (half,reg r); reduce over 16 lanes.
#pragma unroll
        for (int r = 0; r < 8; ++r) {
            float mx = fmaxf(sc[0][r], sc[1][r]);
            mx = fmaxf(mx, __shfl_xor(mx, 1, 32));
            mx = fmaxf(mx, __shfl_xor(mx, 2, 32));
            mx = fmaxf(mx, __shfl_xor(mx, 4, 32));
            mx = fmaxf(mx, __shfl_xor(mx, 8, 32));
            float mnew = fmaxf(mi[r], mx);
            float alph = expf(mi[r] - mnew);
            float p0 = expf(sc[0][r] - mnew);
            float p1 = expf(sc[1][r] - mnew);
            float sr = p0 + p1;
            sr += __shfl_xor(sr, 1, 32);
            sr += __shfl_xor(sr, 2, 32);
            sr += __shfl_xor(sr, 4, 32);
            sr += __shfl_xor(sr, 8, 32);
            li[r] = li[r] * alph + sr;
            mi[r] = mnew;
#pragma unroll
            for (int t = 0; t < 4; ++t) o[t][r] *= alph;
            sP[half * 8 + r][l15]      = (_Float16)p0;
            sP[half * 8 + r][16 + l15] = (_Float16)p1;
        }
        __syncthreads();

        // P (16x32) as A-fragment, V^T as B-fragment: O += P @ V.
        v16h pf;
        ((h8*)&pf)[0] = *(const h8*)&sP[l15][half * 16];
        ((h8*)&pf)[1] = *(const h8*)&sP[l15][half * 16 + 8];
#pragma unroll
        for (int t = 0; t < 4; ++t) {
            v16h vf = *(const v16h*)&sVt[t * 16 + l15][half * 16];
            o[t] = WMMA_F16(pf, vf, o[t]);
        }
        __syncthreads();
    }

#pragma unroll
    for (int t = 0; t < 4; ++t) {
#pragma unroll
        for (int r = 0; r < 8; ++r) {
            int row = q0 + half * 8 + r;
            O[base + (size_t)row * E + t * 16 + l15] = (_Float16)(o[t][r] / li[r]);
        }
    }
}

// ---------------------------------------------------------------------------
// LayerNorm: one wave per row of E=512. Emits fp32 (+optional fp16) outputs.
// ---------------------------------------------------------------------------
__global__ __launch_bounds__(32) void layernorm_k(
    const float* __restrict__ X, const float* __restrict__ g,
    const float* __restrict__ bta, float* __restrict__ outF,
    _Float16* __restrict__ outH, int E) {
    const int row = blockIdx.x, lane = threadIdx.x;
    const float* x = X + (size_t)row * E;
    float s = 0.f, ss = 0.f;
    for (int i = lane; i < E; i += 32) { float v = x[i]; s += v; ss += v * v; }
#pragma unroll
    for (int m = 1; m < 32; m <<= 1) {
        s  += __shfl_xor(s, m, 32);
        ss += __shfl_xor(ss, m, 32);
    }
    const float mu   = s / (float)E;
    const float var  = ss / (float)E - mu * mu;
    const float rstd = rsqrtf(var + 1e-5f);
    for (int i = lane; i < E; i += 32) {
        float v = (x[i] - mu) * rstd * g[i] + bta[i];
        if (outF) outF[(size_t)row * E + i] = v;
        if (outH) outH[(size_t)row * E + i] = (_Float16)v;
    }
}

// ---------------------------------------------------------------------------
// Host orchestration
// ---------------------------------------------------------------------------
extern "C" void kernel_launch(void* const* d_in, const int* in_sizes, int n_in,
                              void* d_out, int out_size, void* d_ws, size_t ws_size,
                              hipStream_t stream) {
    (void)in_sizes; (void)n_in; (void)out_size; (void)ws_size;
    const int Bb = 2, Nn = 2048, E = 512, H = 8, F = 2048;
    const int T = Bb * Nn;  // 4096 rows

    const float* x   = (const float*)d_in[0];
    const float* enc = (const float*)d_in[1];
    const float* sa_wq = (const float*)d_in[2];  const float* sa_bq = (const float*)d_in[3];
    const float* sa_wk = (const float*)d_in[4];  const float* sa_bk = (const float*)d_in[5];
    const float* sa_wv = (const float*)d_in[6];  const float* sa_bv = (const float*)d_in[7];
    const float* sa_wo = (const float*)d_in[8];  const float* sa_bo = (const float*)d_in[9];
    const float* ca_wq = (const float*)d_in[10]; const float* ca_bq = (const float*)d_in[11];
    const float* ca_wk = (const float*)d_in[12]; const float* ca_bk = (const float*)d_in[13];
    const float* ca_wv = (const float*)d_in[14]; const float* ca_bv = (const float*)d_in[15];
    const float* ca_wo = (const float*)d_in[16]; const float* ca_bo = (const float*)d_in[17];
    const float* ln1_g = (const float*)d_in[18]; const float* ln1_b = (const float*)d_in[19];
    const float* ln2_g = (const float*)d_in[20]; const float* ln2_b = (const float*)d_in[21];
    const float* ln3_g = (const float*)d_in[22]; const float* ln3_b = (const float*)d_in[23];
    const float* ff_w1 = (const float*)d_in[24]; const float* ff_b1 = (const float*)d_in[25];
    const float* ff_w2 = (const float*)d_in[26]; const float* ff_b2 = (const float*)d_in[27];

    // Workspace carve-up (~96 MiB total).
    char* ws = (char*)d_ws;
    size_t off = 0;
    auto alloc = [&](size_t bytes) -> void* {
        void* p = ws + off;
        off = (off + bytes + 255) & ~(size_t)255;
        return p;
    };
    _Float16* xh   = (_Float16*)alloc((size_t)T * E * 2);
    _Float16* ench = (_Float16*)alloc((size_t)T * E * 2);
    _Float16* wt[8];
    for (int i = 0; i < 8; ++i) wt[i] = (_Float16*)alloc((size_t)E * E * 2);
    _Float16* wt_ff1 = (_Float16*)alloc((size_t)F * E * 2);  // [N=2048][K=512]
    _Float16* wt_ff2 = (_Float16*)alloc((size_t)E * F * 2);  // [N=512][K=2048]
    _Float16* Qh  = (_Float16*)alloc((size_t)T * E * 2);
    _Float16* Kh  = (_Float16*)alloc((size_t)T * E * 2);
    _Float16* Vh  = (_Float16*)alloc((size_t)T * E * 2);
    _Float16* Oh  = (_Float16*)alloc((size_t)T * E * 2);
    float*    fY1 = (float*)alloc((size_t)T * E * 4);
    float*    fX1 = (float*)alloc((size_t)T * E * 4);
    _Float16* hX1 = (_Float16*)alloc((size_t)T * E * 2);
    float*    fY2 = (float*)alloc((size_t)T * E * 4);
    float*    fX2 = (float*)alloc((size_t)T * E * 4);
    _Float16* hX2 = (_Float16*)alloc((size_t)T * E * 2);
    _Float16* hFF = (_Float16*)alloc((size_t)T * F * 2);
    float*    fY3 = (float*)alloc((size_t)T * E * 4);

    // --- precision conversion + weight pre-transpose ---
    {
        int n = T * E;
        cast_f16_k<<<(n + 255) / 256, 256, 0, stream>>>(x, xh, n);
        cast_f16_k<<<(n + 255) / 256, 256, 0, stream>>>(enc, ench, n);
    }
    const float* Wsrc[8] = {sa_wq, sa_wk, sa_wv, sa_wo, ca_wq, ca_wk, ca_wv, ca_wo};
    for (int i = 0; i < 8; ++i)
        transpose_cast_k<<<dim3(E / 32, E / 32), 256, 0, stream>>>(Wsrc[i], wt[i], E, E);
    transpose_cast_k<<<dim3(F / 32, E / 32), 256, 0, stream>>>(ff_w1, wt_ff1, E, F);
    transpose_cast_k<<<dim3(E / 32, F / 32), 256, 0, stream>>>(ff_w2, wt_ff2, F, E);

    const dim3 gNE(E / BN, T / BM);   // N=512 output
    const dim3 gNF(F / BN, T / BM);   // N=2048 output
    const float qscale = 0.125f;      // 1/sqrt(DH=64), folded into Q projection
    const int attnBlocks = Bb * H * (Nn / 16);

    // --- self attention ---
    gemm_wmma_k<<<gNE, 256, 0, stream>>>(xh, wt[0], sa_bq, nullptr, nullptr, Qh, T, E, E, qscale, 0);
    gemm_wmma_k<<<gNE, 256, 0, stream>>>(xh, wt[1], sa_bk, nullptr, nullptr, Kh, T, E, E, 1.0f, 0);
    gemm_wmma_k<<<gNE, 256, 0, stream>>>(xh, wt[2], sa_bv, nullptr, nullptr, Vh, T, E, E, 1.0f, 0);
    attn_wmma_k<<<attnBlocks, 32, 0, stream>>>(Qh, Kh, Vh, Oh, Nn, H, 1);
    gemm_wmma_k<<<gNE, 256, 0, stream>>>(Oh, wt[3], sa_bo, x, fY1, nullptr, T, E, E, 1.0f, 0);
    layernorm_k<<<T, 32, 0, stream>>>(fY1, ln1_g, ln1_b, fX1, hX1, E);

    // --- cross attention ---
    gemm_wmma_k<<<gNE, 256, 0, stream>>>(hX1, wt[4], ca_bq, nullptr, nullptr, Qh, T, E, E, qscale, 0);
    gemm_wmma_k<<<gNE, 256, 0, stream>>>(ench, wt[5], ca_bk, nullptr, nullptr, Kh, T, E, E, 1.0f, 0);
    gemm_wmma_k<<<gNE, 256, 0, stream>>>(ench, wt[6], ca_bv, nullptr, nullptr, Vh, T, E, E, 1.0f, 0);
    attn_wmma_k<<<attnBlocks, 32, 0, stream>>>(Qh, Kh, Vh, Oh, Nn, H, 0);
    gemm_wmma_k<<<gNE, 256, 0, stream>>>(Oh, wt[7], ca_bo, fX1, fY2, nullptr, T, E, E, 1.0f, 0);
    layernorm_k<<<T, 32, 0, stream>>>(fY2, ln2_g, ln2_b, fX2, hX2, E);

    // --- feed forward ---
    gemm_wmma_k<<<gNF, 256, 0, stream>>>(hX2, wt_ff1, ff_b1, nullptr, nullptr, hFF, T, E, F, 1.0f, 1);
    gemm_wmma_k<<<gNE, 256, 0, stream>>>(hFF, wt_ff2, ff_b2, fX2, fY3, nullptr, T, F, E, 1.0f, 0);
    layernorm_k<<<T, 32, 0, stream>>>(fY3, ln3_g, ln3_b, (float*)d_out, nullptr, E);
}
